// MultiHeadAttention_46351287059124
// MI455X (gfx1250) — matrix operand order repaired and checked by reading, not compiled, not run on hardware
//
#include <hip/hip_runtime.h>
#include <stdint.h>

// ---------------------------------------------------------------------------
// Problem constants (from reference)
// ---------------------------------------------------------------------------
#define D_MODEL   1024
#define NUM_HEADS 16
#define D_K       64
#define BATCH     4
#define SEQ       2048
#define MTOT      (BATCH * SEQ)     // 8192 tokens

// ---------------------------------------------------------------------------
// Types for CDNA5 WMMA (wave32, 16x16x32 bf16 -> f32)
// ---------------------------------------------------------------------------
typedef __bf16 bf16_t;
typedef __attribute__((ext_vector_type(16))) __bf16        v16bf;
typedef __attribute__((ext_vector_type(8)))  float         v8f;
typedef __attribute__((ext_vector_type(4)))  unsigned int  u32x4;
typedef __attribute__((ext_vector_type(2)))  unsigned int  u32x2;
typedef __attribute__((ext_vector_type(4)))  float         f32x4;

union FragB16 { v16bf v; u32x4 q[2]; };

static __device__ __forceinline__ unsigned short f2bf(float f) {
    union { float f; unsigned u; } x; x.f = f;
    unsigned r = x.u + 0x7FFFu + ((x.u >> 16) & 1u);   // round-to-nearest-even
    return (unsigned short)(r >> 16);
}

static __device__ __forceinline__ v8f wmma_bf16(const FragB16& a, const FragB16& b, v8f c) {
    // D = A(16x32 bf16) * B(32x16 bf16) + C(16x16 f32)
    return __builtin_amdgcn_wmma_f32_16x16x32_bf16(false, a.v, false, b.v,
                                                   (short)0, c, false, false);
}

// ---------------------------------------------------------------------------
// f32 -> bf16 conversion (vectorized, packed b64 stores)
// ---------------------------------------------------------------------------
__global__ void cvt_f32_bf16(const float* __restrict__ in,
                             unsigned short* __restrict__ out, int n) {
    int i = (blockIdx.x * blockDim.x + threadIdx.x) * 4;
    if (i < n) {
        f32x4 v = *reinterpret_cast<const f32x4*>(in + i);
        u32x2 o;
        o[0] = (unsigned)f2bf(v[0]) | ((unsigned)f2bf(v[1]) << 16);
        o[1] = (unsigned)f2bf(v[2]) | ((unsigned)f2bf(v[3]) << 16);
        *reinterpret_cast<u32x2*>(out + i) = o;
    }
}

// ---------------------------------------------------------------------------
// GEMM: Y = A @ W^T   (A:[M,K] bf16 row-major, W:[N,K] bf16 row-major)
// MODE 0: Y bf16 [M,N] ; MODE 1: Y bf16 transposed [N,M] ; MODE 2: Y f32 [M,N]
// 4 waves/block; each wave owns a 32x64 tile: per k-step 2 A-frags + 4 B-frags
// feed 8 WMMAs (each B-fragment reused by both row tiles from registers).
// ---------------------------------------------------------------------------
template<int MODE>
__global__ __launch_bounds__(128, 1) void gemm_bf16_wmma(
    const unsigned short* __restrict__ A,
    const unsigned short* __restrict__ W,
    void* __restrict__ Y, int M, int N, int K)
{
    const int lane = threadIdx.x & 31;
    const int wave = threadIdx.x >> 5;
    const int hi16 = lane >> 4;          // 0: lanes 0-15, 1: lanes 16-31
    const int l16  = lane & 15;

    const int mbase = blockIdx.y * 128 + wave * 32;   // 32 rows per wave
    const int nbase = blockIdx.x * 64;

    const int kbA = hi16 ? 8 : 0;        // A-matrix k-half per ISA layout
    const int kbB = hi16 ? 8 : 0;        // B-matrix k-offset per ISA layout

    v8f zero = {};
    v8f acc[2][4];
#pragma unroll
    for (int qt = 0; qt < 2; ++qt)
#pragma unroll
        for (int t = 0; t < 4; ++t) acc[qt][t] = zero;

    const unsigned short* arow0 = A + (size_t)(mbase + l16) * K;
    const unsigned short* arow1 = arow0 + (size_t)16 * K;

    for (int k = 0; k < K; k += 32) {
        FragB16 a0, a1;
        {
            const unsigned short* ap0 = arow0 + k + kbA;
            const unsigned short* ap1 = arow1 + k + kbA;
            a0.q[0] = *reinterpret_cast<const u32x4*>(ap0);        // K = kbA+0..7
            a0.q[1] = *reinterpret_cast<const u32x4*>(ap0 + 16);   // K = kbA+16..23
            a1.q[0] = *reinterpret_cast<const u32x4*>(ap1);
            a1.q[1] = *reinterpret_cast<const u32x4*>(ap1 + 16);
        }
#pragma unroll
        for (int t = 0; t < 4; ++t) {
            FragB16 b;
            const unsigned short* wp = W + (size_t)(nbase + t * 16 + l16) * K + k + kbB;
            b.q[0] = *reinterpret_cast<const u32x4*>(wp);      // K = kbB+0..7
            b.q[1] = *reinterpret_cast<const u32x4*>(wp + 16); // K = kbB+16..23
            acc[0][t] = wmma_bf16(a0, b, acc[0][t]);
            acc[1][t] = wmma_bf16(a1, b, acc[1][t]);
        }
    }

#pragma unroll
    for (int qt = 0; qt < 2; ++qt) {
#pragma unroll
        for (int t = 0; t < 4; ++t) {
            const int n = nbase + t * 16 + l16;
#pragma unroll
            for (int r = 0; r < 8; ++r) {
                const int m = mbase + qt * 16 + r + (hi16 ? 8 : 0);
                float val = acc[qt][t][r];
                if (MODE == 0)      ((unsigned short*)Y)[(size_t)m * N + n] = f2bf(val);
                else if (MODE == 1) ((unsigned short*)Y)[(size_t)n * M + m] = f2bf(val);
                else                ((float*)Y)[(size_t)m * N + n] = val;
            }
        }
    }
}

// ---------------------------------------------------------------------------
// Flash attention: per wave two 16-query tiles (32 q rows) of one (b,h);
// keys streamed in chunks of 32 with online softmax in exp2 domain.
// Row-sum reduction is deferred: per-lane partial l, one butterfly at the end.
// Q,K bf16 [MTOT,D_MODEL]; V pre-transposed bf16 [D_MODEL,MTOT]; O bf16.
// ---------------------------------------------------------------------------
__global__ __launch_bounds__(128, 1) void attn_kernel(
    const unsigned short* __restrict__ Q,
    const unsigned short* __restrict__ Km,
    const unsigned short* __restrict__ Vt,
    unsigned short* __restrict__ O)
{
    __shared__ __align__(16) unsigned short Plds[4][2][16 * 40]; // per wave/qtile P (pad 40)

    const int lane = threadIdx.x & 31;
    const int wave = threadIdx.x >> 5;
    const int hi16 = lane >> 4;
    const int l16  = lane & 15;

    const int bh    = blockIdx.x;
    const int b     = bh / NUM_HEADS;
    const int h     = bh % NUM_HEADS;
    const int qbase = blockIdx.y * 128 + wave * 32;   // 32 q rows per wave
    const int hcol  = h * D_K;

    const int kbA = hi16 ? 8 : 0;
    const int kbB = hi16 ? 8 : 0;

    // Q A-fragments: [qtile][k-step over D_K=64]
    FragB16 aQ[2][2];
#pragma unroll
    for (int qt = 0; qt < 2; ++qt) {
        const unsigned short* qp =
            Q + (size_t)(b * SEQ + qbase + qt * 16 + l16) * D_MODEL + hcol + kbA;
#pragma unroll
        for (int s = 0; s < 2; ++s) {
            aQ[qt][s].q[0] = *reinterpret_cast<const u32x4*>(qp + s * 32);
            aQ[qt][s].q[1] = *reinterpret_cast<const u32x4*>(qp + s * 32 + 16);
        }
    }

    v8f zero = {};
    v8f o[2][4];
    float mrow[2][8], lpart[2][8];
#pragma unroll
    for (int qt = 0; qt < 2; ++qt) {
#pragma unroll
        for (int r = 0; r < 8; ++r) { mrow[qt][r] = -3.0e38f; lpart[qt][r] = 0.0f; }
#pragma unroll
        for (int t = 0; t < 4; ++t) o[qt][t] = zero;
    }

    // exp2 domain: exp(x/sqrt(dk)) == exp2(x * 0.125 * log2(e))
    const float SC = 0.125f * 1.44269504088896340736f;

    for (int c = 0; c < SEQ; c += 32) {
        // ---- scores: S[qt] = Q(16x64) * K^T(64x32), shared K fragments ----
        const unsigned short* kp0 = Km + (size_t)(b * SEQ + c + l16) * D_MODEL + hcol + kbB;
        const unsigned short* kp1 = kp0 + (size_t)16 * D_MODEL;
        if (c + 32 < SEQ) {  // CDNA5 global_prefetch_b8 on next chunk
            __builtin_prefetch((const void*)(kp0 + (size_t)32 * D_MODEL), 0, 3);
            __builtin_prefetch((const void*)(Vt + (size_t)(hcol + l16) * MTOT + b * SEQ + c + 32), 0, 3);
        }
        v8f s0[2] = { zero, zero }, s1[2] = { zero, zero };
#pragma unroll
        for (int s = 0; s < 2; ++s) {
            FragB16 bk0, bk1;
            bk0.q[0] = *reinterpret_cast<const u32x4*>(kp0 + s * 32);
            bk0.q[1] = *reinterpret_cast<const u32x4*>(kp0 + s * 32 + 16);
            bk1.q[0] = *reinterpret_cast<const u32x4*>(kp1 + s * 32);
            bk1.q[1] = *reinterpret_cast<const u32x4*>(kp1 + s * 32 + 16);
#pragma unroll
            for (int qt = 0; qt < 2; ++qt) {
                s0[qt] = wmma_bf16(aQ[qt][s], bk0, s0[qt]);
                s1[qt] = wmma_bf16(aQ[qt][s], bk1, s1[qt]);
            }
        }

        // ---- online softmax; only the max needs a per-chunk butterfly ----
#pragma unroll
        for (int qt = 0; qt < 2; ++qt) {
            unsigned short* pw = &Plds[wave][qt][0];
#pragma unroll
            for (int r = 0; r < 8; ++r) {
                float a0 = s0[qt][r] * SC;
                float a1 = s1[qt][r] * SC;
                float cm = fmaxf(a0, a1);
#pragma unroll
                for (int msk = 1; msk < 16; msk <<= 1)
                    cm = fmaxf(cm, __shfl_xor(cm, msk, 32));
                float mn    = fmaxf(mrow[qt][r], cm);
                float alpha = exp2f(mrow[qt][r] - mn);
                mrow[qt][r] = mn;
                float e0 = exp2f(a0 - mn);
                float e1 = exp2f(a1 - mn);
                // deferred row-sum: per-lane partial over this lane's 2 columns
                lpart[qt][r] = lpart[qt][r] * alpha + (e0 + e1);
#pragma unroll
                for (int t = 0; t < 4; ++t) o[qt][t][r] *= alpha;
                int row = r + (hi16 ? 8 : 0);
                pw[row * 40 + l16]      = f2bf(e0);   // keys 0..15
                pw[row * 40 + 16 + l16] = f2bf(e1);   // keys 16..31
            }
        }

        // ---- O[qt] += P(16x32) * V(32x64): V^T gives contiguous B-frags ----
        FragB16 aP[2];
#pragma unroll
        for (int qt = 0; qt < 2; ++qt) {
            const unsigned short* pr = &Plds[wave][qt][0] + l16 * 40 + kbA;
            aP[qt].q[0] = *reinterpret_cast<const u32x4*>(pr);       // keys kbA+0..7
            aP[qt].q[1] = *reinterpret_cast<const u32x4*>(pr + 16);  // keys kbA+16..23
        }
#pragma unroll
        for (int t = 0; t < 4; ++t) {
            FragB16 bv;
            const unsigned short* vp =
                Vt + (size_t)(hcol + t * 16 + l16) * MTOT + b * SEQ + c + kbB;
            bv.q[0] = *reinterpret_cast<const u32x4*>(vp);
            bv.q[1] = *reinterpret_cast<const u32x4*>(vp + 16);
            o[0][t] = wmma_bf16(aP[0], bv, o[0][t]);
            o[1][t] = wmma_bf16(aP[1], bv, o[1][t]);
        }
    }

    // ---- final: one butterfly for the deferred row sums, normalize, store ----
#pragma unroll
    for (int qt = 0; qt < 2; ++qt) {
#pragma unroll
        for (int r = 0; r < 8; ++r) {
            float ls = lpart[qt][r];
#pragma unroll
            for (int msk = 1; msk < 16; msk <<= 1)
                ls += __shfl_xor(ls, msk, 32);
            float inv = 1.0f / ls;
            int row = qbase + qt * 16 + r + (hi16 ? 8 : 0);
#pragma unroll
            for (int t = 0; t < 4; ++t) {
                O[(size_t)(b * SEQ + row) * D_MODEL + hcol + t * 16 + l16] =
                    f2bf(o[qt][t][r] * inv);
            }
        }
    }
}

// ---------------------------------------------------------------------------
// Launch
// ---------------------------------------------------------------------------
extern "C" void kernel_launch(void* const* d_in, const int* in_sizes, int n_in,
                              void* d_out, int out_size, void* d_ws, size_t ws_size,
                              hipStream_t stream) {
    const float* x  = (const float*)d_in[0];
    const float* Wq = (const float*)d_in[1];
    const float* Wk = (const float*)d_in[2];
    const float* Wv = (const float*)d_in[3];
    const float* Wo = (const float*)d_in[4];

    const size_t MB = 1024ull * 1024ull;
    if (ws_size < 72 * MB) return;  // need 72 MB of scratch

    char* ws = (char*)d_ws;
    unsigned short* Xbf = (unsigned short*)(ws + 0);        // 16 MB (reused as O)
    unsigned short* Wqb = (unsigned short*)(ws + 16 * MB);  // 2 MB
    unsigned short* Wkb = (unsigned short*)(ws + 18 * MB);  // 2 MB
    unsigned short* Wvb = (unsigned short*)(ws + 20 * MB);  // 2 MB
    unsigned short* Wob = (unsigned short*)(ws + 22 * MB);  // 2 MB
    unsigned short* Qb  = (unsigned short*)(ws + 24 * MB);  // 16 MB
    unsigned short* Kb  = (unsigned short*)(ws + 40 * MB);  // 16 MB
    unsigned short* Vtb = (unsigned short*)(ws + 56 * MB);  // 16 MB (transposed V)
    unsigned short* Ob  = Xbf;                              // attention output reuses Xbf

    const int nX = MTOT * D_MODEL;
    const int nW = D_MODEL * D_MODEL;
    const int thr = 256;

    cvt_f32_bf16<<<dim3((nX / 4 + thr - 1) / thr), thr, 0, stream>>>(x,  Xbf, nX);
    cvt_f32_bf16<<<dim3((nW / 4 + thr - 1) / thr), thr, 0, stream>>>(Wq, Wqb, nW);
    cvt_f32_bf16<<<dim3((nW / 4 + thr - 1) / thr), thr, 0, stream>>>(Wk, Wkb, nW);
    cvt_f32_bf16<<<dim3((nW / 4 + thr - 1) / thr), thr, 0, stream>>>(Wv, Wvb, nW);
    cvt_f32_bf16<<<dim3((nW / 4 + thr - 1) / thr), thr, 0, stream>>>(Wo, Wob, nW);

    dim3 gg(D_MODEL / 64, MTOT / 128);  // (16, 64), 128 threads = 4 waves (32x64/wave)
    gemm_bf16_wmma<0><<<gg, 128, 0, stream>>>(Xbf, Wqb, Qb,  MTOT, D_MODEL, D_MODEL);
    gemm_bf16_wmma<0><<<gg, 128, 0, stream>>>(Xbf, Wkb, Kb,  MTOT, D_MODEL, D_MODEL);
    gemm_bf16_wmma<1><<<gg, 128, 0, stream>>>(Xbf, Wvb, Vtb, MTOT, D_MODEL, D_MODEL);

    attn_kernel<<<dim3(BATCH * NUM_HEADS, SEQ / 128), 128, 0, stream>>>(Qb, Kb, Vtb, Ob);

    gemm_bf16_wmma<2><<<gg, 128, 0, stream>>>(Ob, Wob, d_out, MTOT, D_MODEL, D_MODEL);
}
